// CapsuleLayer_21835613733439
// MI455X (gfx1250) — matrix operand order repaired and checked by reading, not compiled, not run on hardware
//
#include <hip/hip_runtime.h>
#include <hip/hip_bf16.h>

// ---------------------------------------------------------------------------
// Fused CapsuleLayer for MI455X (gfx1250, wave32, WMMA + TDM).
// One workgroup (256 threads = 8 waves) per batch element.
// inputs_hat[b] ([32][64][64]) lives in LDS as bf16 (padded rows, 272KB),
// produced by v_wmma_f32_16x16x32_bf16 and consumed by 3 routing iterations,
// FM / pairwise-MF stats, and a WMMA-based 2-head InteractingLayer.
// x[b] is staged into LDS by the Tensor Data Mover (tensor_load_to_lds).
// ---------------------------------------------------------------------------

typedef __bf16 bf16;
typedef __attribute__((ext_vector_type(16))) __bf16 v16bf;
typedef __attribute__((ext_vector_type(8)))  float  v8f;
typedef __attribute__((ext_vector_type(4)))  unsigned int u32x4;
typedef __attribute__((ext_vector_type(8)))  int i32x8;
typedef __attribute__((ext_vector_type(4)))  int i32x4;

#define B_SZ   1024
#define F_INN  64
#define D_INN  64
#define N_CAPS 32
#define D_CAPSZ 64
#define NCOLS  2048           // N_CAPS * D_CAPS
#define E_DIM  128            // HEADS * D_CAPS
#define P_PAIRS 496
#define EPSV   1e-7f

#define IH_STRIDE 68          // padded K-row stride (elements) -> 136B = 34 banks

// ---- LDS layout (bytes) ----------------------------------------------------
#define IH_OFF    0
#define IH_BYTES  (N_CAPS*F_INN*IH_STRIDE*2)     // 278528
#define XA_OFF    (IH_OFF + IH_BYTES)            // x[b] as bf16, padded rows
#define XA_BYTES  (F_INN*IH_STRIDE*2)            // 8704
#define BLOG_OFF  (XA_OFF + XA_BYTES)            // f32 [32][64]
#define CNF_OFF   (BLOG_OFF + 8192)              // f32 [32][64]
#define OUTV_OFF  (CNF_OFF  + 8192)              // f32 [32][64]
#define FWV_OFF   (OUTV_OFF + 8192)              // f32 [32][64]
#define SCL_OFF   (FWV_OFF  + 8192)              // f32 [32]
#define HIFM_OFF  (SCL_OFF  + 128)               // f32 [64]
#define HMF_OFF   (HIFM_OFF + 256)               // f32 [64]
#define HINT_OFF  (HMF_OFF  + 256)               // f32 [128]
#define SMEM_BYTES (HINT_OFF + 512)              // 321152 <= 320KB LDS cap

// TDM staging buffer for x[b] (f32 [64][64] = 16KB) aliases the start of IH;
// it is consumed (converted to XA) before the GEMM overwrites IH.
#define XRAW_OFF  0

// attention-phase buffers alias the (dead) IH region
#define FWVB_OFF  0                              // bf16 [32][68]
#define QB_OFF    4352                           // bf16 [2][32][68]
#define KB_OFF    13056                          // bf16 [2][32][68]
#define VT_OFF    21760                          // bf16 [2][64][36]  (v transposed)
#define ATP_OFF   30976                          // bf16 [2][32][36]  (attn probs)
#define RESF_OFF  35584                          // f32  [32][128]
#define SCR_OFF   51968                          // f32  [2][32][32]

// ws layout: bf16 Wt_main[2048][64] then 4x bf16 Wt_{q,k,v,res}[128][64]
#define WS_QOFF   (NCOLS*64)

__device__ __forceinline__ float bf2f(unsigned int u) {
  union { unsigned int i; float f; } x; x.i = u << 16; return x.f;
}

__device__ __forceinline__ v8f wmma_bf16(v16bf a, v16bf b, v8f c) {
  return __builtin_amdgcn_wmma_f32_16x16x32_bf16(
      false, a, false, b, (short)0, c, false, false);
}

// A fragment (16x32 bf16) from LDS [row][K] (padded). Contiguous k order (same as B):
// lane<16: M=lane, K=kb..kb+15; lane>=16: M=lane-16, K=kb+16..kb+31.
__device__ __forceinline__ v16bf loadA16_lds(const bf16* m, int strideEl,
                                             int rowBase, int kBase, int lane) {
  int r = lane & 15, hi = lane >> 4;
  const bf16* p = m + (rowBase + r) * strideEl + kBase + hi * 16;
  union { v16bf v; uint2 q[4]; } u;
  u.q[0] = *(const uint2*)(p);
  u.q[1] = *(const uint2*)(p + 4);
  u.q[2] = *(const uint2*)(p + 8);
  u.q[3] = *(const uint2*)(p + 12);
  return u.v;
}

// B fragment (32x16 bf16) from LDS [col][K] (padded). Contiguous k order (same as A):
// lane<16: N=lane, K=kb..kb+15; lane>=16: N=lane-16, K=kb+16..kb+31.
__device__ __forceinline__ v16bf loadB16_lds(const bf16* m, int strideEl,
                                             int colBase, int kBase, int lane) {
  int c = lane & 15, hi = lane >> 4;
  const bf16* p = m + (colBase + c) * strideEl + kBase + hi * 16;
  union { v16bf v; uint2 q[4]; } u;
  u.q[0] = *(const uint2*)(p);
  u.q[1] = *(const uint2*)(p + 4);
  u.q[2] = *(const uint2*)(p + 8);
  u.q[3] = *(const uint2*)(p + 12);
  return u.v;
}

// B fragment from global bf16 [col][64] row-major (16B-aligned rows).
__device__ __forceinline__ v16bf loadB16_g(const bf16* m, int colBase,
                                           int kBase, int lane) {
  int c = lane & 15, hi = lane >> 4;
  const bf16* p = m + (colBase + c) * 64 + kBase + hi * 16;
  union { v16bf v; uint4 q[2]; } u;
  u.q[0] = *(const uint4*)(p);
  u.q[1] = *(const uint4*)(p + 8);
  return u.v;
}

// Issue a TDM 2-D tile load: GLOBAL[gaddr] (64x64 f32, row stride 64) -> LDS 0.
__device__ __forceinline__ void tdm_load_x(const float* gsrc) {
  unsigned long long ga = (unsigned long long)(const void*)gsrc;
  u32x4 g0;
  g0[0] = 1u;                                      // count=1, user mode
  g0[1] = 0u;                                      // lds_addr = 0 (XRAW_OFF)
  g0[2] = (unsigned int)ga;                        // global_addr[31:0]
  g0[3] = (unsigned int)((ga >> 32) & 0x1ffffffull) | (2u << 30);  // addr_hi|type=2
  // group1: data_size=4B (code 2), tensor 64x64, tile 64x64, dim0_stride=64
  i32x8 g1 = { 0x20000, 64 << 16, 64 << 16, 64 << 16, 64, 64, 0, 0 };
  i32x4 z4 = { 0, 0, 0, 0 };
#if __clang_major__ >= 23
  i32x8 z8 = { 0, 0, 0, 0, 0, 0, 0, 0 };
  __builtin_amdgcn_tensor_load_to_lds(g0, g1, z4, z4, z8, 0);
#else
  __builtin_amdgcn_tensor_load_to_lds(g0, g1, z4, z4, 0);
#endif
  __builtin_amdgcn_s_wait_tensorcnt(0);
}

// ---------------------------------------------------------------------------
// Kernel 0: one-time weight conversion to bf16, B-fragment-friendly transpose.
// ---------------------------------------------------------------------------
__global__ __launch_bounds__(256) void caps_prep(
    const float* __restrict__ W,                 // [64][2048] (d, n*c)
    const float* __restrict__ Wq, const float* __restrict__ Wk,
    const float* __restrict__ Wv, const float* __restrict__ Wr,
    bf16* __restrict__ ws) {
  int idx = blockIdx.x * 256 + threadIdx.x;
  if (idx < NCOLS * 64) {
    int nc = idx & (NCOLS - 1);
    int d  = idx >> 11;
    ws[nc * 64 + d] = (bf16)W[d * NCOLS + nc];  // Wt_main[nc][d]
  } else {
    int r = idx - NCOLS * 64;                   // 0..32767
    int mat = r >> 13;
    int r2  = r & 8191;
    int e   = r2 & 127;
    int d   = r2 >> 7;
    const float* src = (mat == 0) ? Wq : (mat == 1) ? Wk : (mat == 2) ? Wv : Wr;
    ws[WS_QOFF + mat * 8192 + e * 64 + d] = (bf16)src[d * 128 + e];
  }
}

// ---------------------------------------------------------------------------
// Main fused kernel: one block per batch element b.
// ---------------------------------------------------------------------------
__global__ __launch_bounds__(256) void caps_main(
    const float* __restrict__ x,            // [B][64][64]
    const float* __restrict__ routing_init, // [32][64]
    const float* __restrict__ kernel_fm,    // [32]
    const float* __restrict__ bias_fm,      // [64]
    const float* __restrict__ kernel_mf,    // [496]
    const float* __restrict__ bias_mf,      // [64]
    const float* __restrict__ kernel_hi,    // [32]
    const float* __restrict__ bias_hi,      // [128]
    const bf16*  __restrict__ ws,
    float* __restrict__ out_main,           // [B][256]
    float* __restrict__ out_rs) {           // [B][32][64]
  extern __shared__ char smem[];
  bf16*  IH   = (bf16*)(smem + IH_OFF);
  bf16*  XA   = (bf16*)(smem + XA_OFF);
  float* XRAW = (float*)(smem + XRAW_OFF);
  float* BLOG = (float*)(smem + BLOG_OFF);
  float* CNF  = (float*)(smem + CNF_OFF);
  float* OUTV = (float*)(smem + OUTV_OFF);
  float* FWV  = (float*)(smem + FWV_OFF);
  float* SCL  = (float*)(smem + SCL_OFF);
  float* HIFM = (float*)(smem + HIFM_OFF);
  float* HMF  = (float*)(smem + HMF_OFF);
  float* HINT = (float*)(smem + HINT_OFF);
  bf16*  FWVB = (bf16*)(smem + FWVB_OFF);
  bf16*  QB   = (bf16*)(smem + QB_OFF);
  bf16*  KB   = (bf16*)(smem + KB_OFF);
  bf16*  VT   = (bf16*)(smem + VT_OFF);
  bf16*  ATP  = (bf16*)(smem + ATP_OFF);
  float* RESF = (float*)(smem + RESF_OFF);
  float* SCR  = (float*)(smem + SCR_OFF);

  const int t    = threadIdx.x;
  const int lane = t & 31;
  const int wave = t >> 5;
  const int b    = blockIdx.x;

  // ---- Phase 0: TDM-stage x[b] into LDS, init logits + accumulators --------
  if (wave == 0) {
    tdm_load_x(x + (size_t)b * 4096);   // DMA 16KB tile -> LDS[0], wait tensorcnt
  }
  for (int i = t; i < 2048; i += 256) BLOG[i] = routing_init[i];
  if (t < 64) { HIFM[t] = 0.f; HMF[t] = 0.f; }
  if (t < 128) HINT[t] = 0.f;
  // prefetch the L2-resident bf16 weight panel while we wait
  __builtin_prefetch(ws + (size_t)t * 512, 0, 0);
  __syncthreads();                       // XRAW visible to all waves

  // convert staged f32 x[b] to padded bf16 A-panel (XRAW dies after this)
  for (int i = t; i < 4096; i += 256) {
    int f = i >> 6, d = i & 63;
    XA[f * IH_STRIDE + d] = (bf16)XRAW[i];
  }
  __syncthreads();                       // XA ready; IH region free to overwrite

  // ---- Phase 1: inputs_hat GEMM [64 x 2048], K=64, bf16 WMMA -> IH (LDS) ---
  {
    v16bf Af[4][2];
#pragma unroll
    for (int mt = 0; mt < 4; ++mt)
#pragma unroll
      for (int kh = 0; kh < 2; ++kh)
        Af[mt][kh] = loadA16_lds(XA, IH_STRIDE, mt * 16, kh * 32, lane);
    const int hi = lane >> 4;
    for (int nti = 0; nti < 16; ++nti) {
      int nt = wave * 16 + nti;                 // 128 n-col tiles total
      v16bf B0 = loadB16_g(ws, nt * 16, 0, lane);
      v16bf B1 = loadB16_g(ws, nt * 16, 32, lane);
      int n = (nt * 16) >> 6;
      int c = ((nt * 16) & 63) + (lane & 15);
#pragma unroll
      for (int mt = 0; mt < 4; ++mt) {
        v8f acc = {};
        acc = wmma_bf16(Af[mt][0], B0, acc);
        acc = wmma_bf16(Af[mt][1], B1, acc);
#pragma unroll
        for (int j = 0; j < 8; ++j) {
          int f = mt * 16 + j + hi * 8;         // C/D layout: lanes>=16 -> M+8
          IH[(n * 64 + f) * IH_STRIDE + c] = (bf16)acc[j];
        }
      }
    }
  }
  __syncthreads();

  // softmax over capsules (axis n) per field f, BLOG -> CNF
  auto softmax_caps = [&]() {
    if (t < 64) {
      float mx = -3.4e38f;
#pragma unroll
      for (int n = 0; n < 32; ++n) mx = fmaxf(mx, BLOG[n * 64 + t]);
      float e[32]; float s = 0.f;
#pragma unroll
      for (int n = 0; n < 32; ++n) { e[n] = __expf(BLOG[n * 64 + t] - mx); s += e[n]; }
      float inv = 1.f / s;
#pragma unroll
      for (int n = 0; n < 32; ++n) CNF[n * 64 + t] = e[n] * inv;
    }
    __syncthreads();
  };

  // ---- Phase 2: dynamic routing (3 iterations) -----------------------------
  for (int it = 0; it < 3; ++it) {
    softmax_caps();
    // outputs[n][c] = sum_f c[n][f] * ih[n][f][c]   (lanes span c, pairs)
    for (int p = 0; p < 4; ++p) {
      int n = p * 8 + wave;
      int c0 = lane * 2;
      const bf16* ihn = IH + n * 64 * IH_STRIDE;
      float a0 = 0.f, a1 = 0.f;
      for (int f = 0; f < 64; ++f) {
        unsigned int w = *(const unsigned int*)(ihn + f * IH_STRIDE + c0);
        float cv = CNF[n * 64 + f];
        a0 += cv * bf2f(w & 0xffffu);
        a1 += cv * bf2f(w >> 16);
      }
      OUTV[n * 64 + c0] = a0; OUTV[n * 64 + c0 + 1] = a1;
    }
    __syncthreads();
    // squash scale per capsule (fold the scale into the b-update below)
    if (t < 32) {
      float s = EPSV;
      for (int c = 0; c < 64; ++c) { float v = OUTV[t * 64 + c]; s += v * v; }
      SCL[t] = __fsqrt_rn(s) / (0.5f + s);
    }
    __syncthreads();
    // b[n][f] += sum_c squash(o)[n][c] * ih[n][f][c]  (per-thread c-row)
    for (int p = 0; p < 8; ++p) {
      int n = p * 4 + (t >> 6);
      int f = t & 63;
      const bf16* ihr = IH + (n * 64 + f) * IH_STRIDE;
      float sc = SCL[n];
      float acc = 0.f;
      for (int c = 0; c < 64; c += 4) {
        uint2 w = *(const uint2*)(ihr + c);
        acc += OUTV[n * 64 + c + 0] * bf2f(w.x & 0xffffu)
             + OUTV[n * 64 + c + 1] * bf2f(w.x >> 16)
             + OUTV[n * 64 + c + 2] * bf2f(w.y & 0xffffu)
             + OUTV[n * 64 + c + 3] * bf2f(w.y >> 16);
      }
      BLOG[n * 64 + f] += sc * acc;
    }
    __syncthreads();
  }
  softmax_caps();  // final routing coefficients

  // routing_score output
  for (int i = t; i < 2048; i += 256) out_rs[b * 2048 + i] = CNF[i];

  // ---- Phase 3: field sums + FM stats --------------------------------------
  for (int p = 0; p < 4; ++p) {
    int n = p * 8 + wave;
    int c0 = lane * 2;
    const bf16* ihn = IH + n * 64 * IH_STRIDE;
    float s0 = 0, s1 = 0, q0 = 0, q1 = 0;
    for (int f = 0; f < 64; ++f) {
      unsigned int w = *(const unsigned int*)(ihn + f * IH_STRIDE + c0);
      float cv = CNF[n * 64 + f];
      float v0 = cv * bf2f(w & 0xffffu);
      float v1 = cv * bf2f(w >> 16);
      s0 += v0; s1 += v1; q0 += v0 * v0; q1 += v1 * v1;
    }
    FWV[n * 64 + c0] = s0; FWV[n * 64 + c0 + 1] = s1;
    float kf = kernel_fm[n];
    atomicAdd(&HIFM[c0],     (s0 * s0 - q0) * kf);
    atomicAdd(&HIFM[c0 + 1], (s1 * s1 - q1) * kf);
  }
  __syncthreads();   // IH is dead from here on

  // ---- Phase 4: pairwise MF interactions (496 pairs, triu order) -----------
  {
    int c  = t & 63;
    int pg = t >> 6;                       // 4 groups x 124 pairs
    int p0 = pg * 124;
    int i0 = 0, rem = p0, cnt = 31;
    while (rem >= cnt) { rem -= cnt; ++i0; --cnt; }
    int i = i0, j = i0 + 1 + rem;
    float acc = 0.f;
    for (int p = p0; p < p0 + 124; ++p) {
      acc += kernel_mf[p] * FWV[i * 64 + c] * FWV[j * 64 + c];
      if (++j == 32) { ++i; j = i + 1; }
    }
    atomicAdd(&HMF[c], acc);
  }

  // ---- Phase 5: fwv -> bf16 (A operand for the attention GEMMs) ------------
  for (int i = t; i < 2048; i += 256) {
    int n = i >> 6, c = i & 63;
    FWVB[n * IH_STRIDE + c] = (bf16)FWV[i];
  }
  __syncthreads();

  // ---- Phase 6: q/k/v/res = fwv x W_*  (WMMA, 64 tiles, 8 per wave) --------
  {
    v16bf Aq[2][2];
#pragma unroll
    for (int mt = 0; mt < 2; ++mt)
#pragma unroll
      for (int kh = 0; kh < 2; ++kh)
        Aq[mt][kh] = loadA16_lds(FWVB, IH_STRIDE, mt * 16, kh * 32, lane);
    const bf16* WtQ = ws + WS_QOFF;
    const int hi = lane >> 4;
#pragma unroll
    for (int i = 0; i < 8; ++i) {
      int tile = wave * 8 + i;             // mat(2b) | mt(1b) | et(3b)
      int mat = tile >> 4;
      int mt  = (tile >> 3) & 1;
      int et  = tile & 7;
      const bf16* Wm = WtQ + mat * 8192;
      v16bf B0 = loadB16_g(Wm, et * 16, 0, lane);
      v16bf B1 = loadB16_g(Wm, et * 16, 32, lane);
      v8f acc = {};
      acc = wmma_bf16(Aq[mt][0], B0, acc);
      acc = wmma_bf16(Aq[mt][1], B1, acc);
      int e = et * 16 + (lane & 15);
      int h = e >> 6, fd = e & 63;
      if (mat == 0) {
#pragma unroll
        for (int j = 0; j < 8; ++j) { int n = mt * 16 + j + hi * 8;
          QB[(h * 32 + n) * IH_STRIDE + fd] = (bf16)acc[j]; }
      } else if (mat == 1) {
#pragma unroll
        for (int j = 0; j < 8; ++j) { int n = mt * 16 + j + hi * 8;
          KB[(h * 32 + n) * IH_STRIDE + fd] = (bf16)acc[j]; }
      } else if (mat == 2) {
#pragma unroll
        for (int j = 0; j < 8; ++j) { int n = mt * 16 + j + hi * 8;
          VT[(h * 64 + fd) * 36 + n] = (bf16)acc[j]; }   // transposed for B-frag
      } else {
#pragma unroll
        for (int j = 0; j < 8; ++j) { int n = mt * 16 + j + hi * 8;
          RESF[n * 128 + e] = acc[j]; }
      }
    }
  }
  __syncthreads();

  // ---- Phase 7: scores = q . k^T per head (8 tiles, 1 per wave) ------------
  {
    int h  = wave >> 2;
    int qt = (wave >> 1) & 1;
    int kt = wave & 1;
    v16bf a0 = loadA16_lds(QB + h * 32 * IH_STRIDE, IH_STRIDE, qt * 16, 0, lane);
    v16bf a1 = loadA16_lds(QB + h * 32 * IH_STRIDE, IH_STRIDE, qt * 16, 32, lane);
    v16bf b0 = loadB16_lds(KB + h * 32 * IH_STRIDE, IH_STRIDE, kt * 16, 0, lane);
    v16bf b1 = loadB16_lds(KB + h * 32 * IH_STRIDE, IH_STRIDE, kt * 16, 32, lane);
    v8f acc = {};
    acc = wmma_bf16(a0, b0, acc);
    acc = wmma_bf16(a1, b1, acc);
    int hi = lane >> 4;
    int m = kt * 16 + (lane & 15);
#pragma unroll
    for (int j = 0; j < 8; ++j) {
      int n = qt * 16 + j + hi * 8;
      SCR[(h * 32 + n) * 32 + m] = acc[j];
    }
  }
  __syncthreads();

  // ---- Phase 8: softmax over keys -> bf16 attention probs ------------------
  if (t < 64) {
    int h = t >> 5, n = t & 31;
    const float* row = SCR + (h * 32 + n) * 32;
    float mx = row[0];
#pragma unroll
    for (int m = 1; m < 32; ++m) mx = fmaxf(mx, row[m]);
    float e[32]; float s = 0.f;
#pragma unroll
    for (int m = 0; m < 32; ++m) { e[m] = __expf(row[m] - mx); s += e[m]; }
    float inv = 1.f / s;
#pragma unroll
    for (int m = 0; m < 32; ++m) ATP[(h * 32 + n) * 36 + m] = (bf16)(e[m] * inv);
  }
  __syncthreads();

  // ---- Phase 9: out = attn . v, + res, relu, high-order accumulation -------
  {
    const int hi = lane >> 4;
#pragma unroll
    for (int i = 0; i < 2; ++i) {
      int tile = wave * 2 + i;             // h(1b) | mt(1b) | ft(2b)
      int h  = tile >> 3;
      int mt = (tile >> 2) & 1;
      int ft = tile & 3;
      v16bf a  = loadA16_lds(ATP + h * 32 * 36, 36, mt * 16, 0, lane);
      v16bf bb = loadB16_lds(VT + h * 64 * 36, 36, ft * 16, 0, lane);
      v8f acc = {};
      acc = wmma_bf16(a, bb, acc);
      int e = h * 64 + ft * 16 + (lane & 15);
#pragma unroll
      for (int j = 0; j < 8; ++j) {
        int n = mt * 16 + j + hi * 8;
        float v = acc[j] + RESF[n * 128 + e];
        v = fmaxf(v, 0.f);
        atomicAdd(&HINT[e], v * kernel_hi[n]);
      }
    }
  }
  __syncthreads();

  // ---- Phase 10: biases + final stores -------------------------------------
  if (t < 64) {
    out_main[b * 256 + t]      = HIFM[t] + bias_fm[t];
    out_main[b * 256 + 64 + t] = HMF[t] + bias_mf[t];
  }
  if (t < 128) out_main[b * 256 + 128 + t] = HINT[t] + bias_hi[t];
}

// ---------------------------------------------------------------------------
extern "C" void kernel_launch(void* const* d_in, const int* in_sizes, int n_in,
                              void* d_out, int out_size, void* d_ws, size_t ws_size,
                              hipStream_t stream) {
  const float* x   = (const float*)d_in[0];
  const float* W   = (const float*)d_in[1];
  const float* ri  = (const float*)d_in[2];
  const float* kfm = (const float*)d_in[3];
  const float* bfm = (const float*)d_in[4];
  const float* kmf = (const float*)d_in[5];
  const float* bmf = (const float*)d_in[6];
  const float* khi = (const float*)d_in[7];
  const float* bhi = (const float*)d_in[8];
  const float* Wq  = (const float*)d_in[9];
  const float* Wk  = (const float*)d_in[10];
  const float* Wv  = (const float*)d_in[11];
  const float* Wr  = (const float*)d_in[12];
  float* outm  = (float*)d_out;
  float* outrs = outm + B_SZ * 256;        // tuple element 2: routing_score
  bf16*  ws    = (bf16*)d_ws;              // needs 320KB of workspace

  hipFuncSetAttribute((const void*)caps_main,
                      hipFuncAttributeMaxDynamicSharedMemorySize, SMEM_BYTES);

  // one-time per launch: bf16-transposed weights ((2048+4*128) x 64)
  caps_prep<<<640, 256, 0, stream>>>(W, Wq, Wk, Wv, Wr, ws);
  caps_main<<<B_SZ, 256, SMEM_BYTES, stream>>>(x, ri, kfm, bfm, kmf, bmf,
                                               khi, bhi, ws, outm, outrs);
}
